// DeepLPF_90005334655040
// MI455X (gfx1250) — matrix operand order repaired and checked by reading, not compiled, not run on hardware
//
#include <hip/hip_runtime.h>
#include <math.h>

typedef _Float16 half_t;
typedef __attribute__((ext_vector_type(16))) _Float16 v16h;
typedef __attribute__((ext_vector_type(8)))  _Float16 v8h;
typedef __attribute__((ext_vector_type(8)))  float    v8f;

#define EPS_F 1e-10f
#define PI_F  3.14159265358979f

// ---------------------------------------------------------------------------
// Weight prep: f32 HWIO -> f16 [co][k] (k = tap*Cin + ci), zero padded.
// ---------------------------------------------------------------------------
__global__ void prep_bb_kernel(const float* __restrict__ bb_w, half_t* __restrict__ wt) {
    int idx = blockIdx.x * blockDim.x + threadIdx.x;   // [64 co][32 k]
    if (idx >= 64 * 32) return;
    int co = idx >> 5, k = idx & 31;
    float v = 0.f;
    if (co < 61 && k < 27) v = bb_w[k * 61 + co];      // (3,3,3,61): k = tap*3+ci
    wt[idx] = (half_t)v;
}

__global__ void prep_head_kernel(const float* __restrict__ w0,
                                 const float* __restrict__ w1,
                                 const float* __restrict__ w2,
                                 half_t* __restrict__ wt) {
    // wt: [head][layer][co:64][k:576], k = tap*64 + ci
    long idx = (long)blockIdx.x * blockDim.x + threadIdx.x;
    const long per_head = 4L * 64 * 576;
    if (idx >= 3 * per_head) return;
    int  h  = (int)(idx / per_head);
    long r  = idx % per_head;
    int  l  = (int)(r / (64 * 576));
    int  r2 = (int)(r % (64 * 576));
    int  co = r2 / 576, k = r2 % 576;
    int  tap = k >> 6, ci = k & 63;
    const float* w = (h == 0) ? w0 : ((h == 1) ? w1 : w2);
    // conv_w (4,3,3,64,64): (((l*9+tap)*64)+ci)*64 + co
    wt[idx] = (half_t)w[(((long)l * 9 + tap) * 64 + ci) * 64 + co];
}

// ---------------------------------------------------------------------------
// Backbone 3x3 conv (3->61, stride 1, SAME) + ReLU, fused concat(img) -> fi f16.
// WMMA 16x16x32 f16: one K chunk (27 real K, padded to 32).
// ---------------------------------------------------------------------------
__global__ void __launch_bounds__(128)
backbone_kernel(const float* __restrict__ img, const half_t* __restrict__ wt,
                const float* __restrict__ bias, half_t* __restrict__ fi) {
    const int H = 512, W = 512;
    __shared__ half_t Asm[16 * 32];
    int tid = threadIdx.x;
    const int tilesPerB = (H * W) / 16;                 // 16384
    int b = blockIdx.x / tilesPerB;
    int pbase = (blockIdx.x % tilesPerB) * 16;

    for (int i = tid; i < 16 * 32; i += 128) Asm[i] = (half_t)0.f;
    __syncthreads();
    for (int t = tid; t < 16 * 27; t += 128) {
        int pix = t / 27, r = t % 27;
        int tap = r / 3, ci = r % 3;
        int p  = pbase + pix;
        int oy = p / W, ox = p % W;
        int iy = oy - 1 + tap / 3, ix = ox - 1 + tap % 3;
        float v = 0.f;
        if (iy >= 0 && iy < H && ix >= 0 && ix < W)
            v = img[(((long)b * H + iy) * W + ix) * 3 + ci];
        Asm[pix * 32 + r] = (half_t)v;
    }
    __syncthreads();

    int wave = tid >> 5, lane = tid & 31;
    int row = lane & 15, hiHalf = lane >> 4;
    int co = wave * 16 + row;

    v16h afrag = {};
    {
        v8h l0 = *(const v8h*)&Asm[row * 32 + hiHalf * 8];
        v8h h0 = *(const v8h*)&Asm[row * 32 + 16 + hiHalf * 8];
        for (int i = 0; i < 8; i++) { afrag[i] = l0[i]; afrag[8 + i] = h0[i]; }
    }
    v16h bfrag;
    {
        v8h bl = *(const v8h*)(wt + co * 32 + hiHalf * 8);
        v8h bh = *(const v8h*)(wt + co * 32 + 16 + hiHalf * 8);
        for (int i = 0; i < 8; i++) { bfrag[i] = bl[i]; bfrag[8 + i] = bh[i]; }
    }
    v8f acc = {};
    acc = __builtin_amdgcn_wmma_f32_16x16x32_f16(false, afrag, false, bfrag,
                                                 (short)0, acc, false, false);
    float bv = (co < 61) ? bias[co] : 0.f;
    for (int r8 = 0; r8 < 8; r8++) {
        int p  = pbase + r8 + hiHalf * 8;
        int oy = p / W, ox = p % W;
        float v = fmaxf(acc[r8] + bv, 0.f);             // ReLU
        if (co >= 61)                                   // concat img channels
            v = img[(((long)b * H + oy) * W + ox) * 3 + (co - 61)];
        fi[(((long)b * H + oy) * W + ox) * 64 + co] = (half_t)v;
    }
}

// ---------------------------------------------------------------------------
// Bilinear resize fi (B,512,512,64) f16 -> x0 (B,300,300,64) f16.
// ---------------------------------------------------------------------------
__global__ void resize_kernel(const half_t* __restrict__ fi, half_t* __restrict__ x0) {
    long idx = (long)blockIdx.x * blockDim.x + threadIdx.x;
    const long total = 4L * 300 * 300 * 64;
    if (idx >= total) return;
    int c = (int)(idx & 63);
    long r = idx >> 6;
    int ox = (int)(r % 300); r /= 300;
    int oy = (int)(r % 300); int b = (int)(r / 300);
    const float scale = 512.f / 300.f;
    float fy = (oy + 0.5f) * scale - 0.5f;
    float fx = (ox + 0.5f) * scale - 0.5f;
    int y0 = (int)floorf(fy), x0i = (int)floorf(fx);
    float wy = fy - y0, wx = fx - x0i;
    int y1 = y0 + 1, x1 = x0i + 1;
    y0 = y0 < 0 ? 0 : (y0 > 511 ? 511 : y0);
    y1 = y1 < 0 ? 0 : (y1 > 511 ? 511 : y1);
    x0i = x0i < 0 ? 0 : (x0i > 511 ? 511 : x0i);
    x1 = x1 < 0 ? 0 : (x1 > 511 ? 511 : x1);
    long base = (long)b * 512 * 512;
    float v00 = (float)fi[((base + (long)y0 * 512 + x0i)) * 64 + c];
    float v01 = (float)fi[((base + (long)y0 * 512 + x1 )) * 64 + c];
    float v10 = (float)fi[((base + (long)y1 * 512 + x0i)) * 64 + c];
    float v11 = (float)fi[((base + (long)y1 * 512 + x1 )) * 64 + c];
    float v = (v00 * (1.f - wx) + v01 * wx) * (1.f - wy)
            + (v10 * (1.f - wx) + v11 * wx) * wy;
    x0[idx] = (half_t)v;
}

// ---------------------------------------------------------------------------
// 3x3 stride-2 conv, 64->64, SAME, leaky ReLU(0.2). Implicit GEMM:
// WG = 128 thr = 4 waves; 32 out-pixels x 64 out-ch per WG.
// Each wave holds TWO accumulators (two 16-pixel M-tiles) sharing one B frag,
// so the inner loop issues 2 WMMAs per B load pair (halves weight traffic).
// K = 9*64 = 576 -> 18 K-chunks, 36 WMMAs per wave.
// ---------------------------------------------------------------------------
__global__ void __launch_bounds__(128)
conv2s_kernel(const half_t* __restrict__ X, const half_t* __restrict__ wt,
              const float* __restrict__ bias, half_t* __restrict__ Y,
              int Hi, int Wi, int Ho, int Wo, int pad) {
    const int ASTR = 584;                               // 576 + pad halves
    __shared__ half_t Asm[32 * ASTR];                   // 37,376 B
    int tid = threadIdx.x;
    int HoWo = Ho * Wo;
    int tilesPerB = (HoWo + 31) >> 5;
    int b = blockIdx.x / tilesPerB;
    int pbase = (blockIdx.x % tilesPerB) * 32;

    // Gather A: 32 pix x 9 taps x 8 parts of 8 halves (16B each).
    for (int t = tid; t < 32 * 9 * 8; t += 128) {
        int pix = t / 72, r = t % 72;
        int tap = r >> 3, part = r & 7;
        int p = pbase + pix;
        v8h v = {};
        if (p < HoWo) {
            int oy = p / Wo, ox = p % Wo;
            int iy = oy * 2 - pad + tap / 3;
            int ix = ox * 2 - pad + tap % 3;
            if (iy >= 0 && iy < Hi && ix >= 0 && ix < Wi)
                v = *(const v8h*)(X + ((((long)b * Hi + iy) * Wi + ix) << 6) + (part << 3));
        }
        *(v8h*)&Asm[pix * ASTR + (tap << 6) + (part << 3)] = v;
    }
    __syncthreads();

    int wave = tid >> 5, lane = tid & 31;
    int row = lane & 15, hiHalf = lane >> 4;
    int co = wave * 16 + row;
    const half_t* wrow = wt + (long)co * 576;
    __builtin_prefetch(wrow, 0, 0);                     // global_prefetch_b8

    v8f acc0 = {}, acc1 = {};
    for (int kb = 0; kb < 576; kb += 32) {
        v16h bfrag;
        {
            v8h bl = *(const v8h*)(wrow + kb + hiHalf * 8);
            v8h bh = *(const v8h*)(wrow + kb + 16 + hiHalf * 8);
            for (int i = 0; i < 8; i++) { bfrag[i] = bl[i]; bfrag[8 + i] = bh[i]; }
        }
        v16h a0 = {}, a1 = {};
        {
            v8h l0 = *(const v8h*)&Asm[row * ASTR + kb + hiHalf * 8];
            v8h h0 = *(const v8h*)&Asm[row * ASTR + kb + 16 + hiHalf * 8];
            for (int i = 0; i < 8; i++) { a0[i] = l0[i]; a0[8 + i] = h0[i]; }
            v8h l1 = *(const v8h*)&Asm[(16 + row) * ASTR + kb + hiHalf * 8];
            v8h h1 = *(const v8h*)&Asm[(16 + row) * ASTR + kb + 16 + hiHalf * 8];
            for (int i = 0; i < 8; i++) { a1[i] = l1[i]; a1[8 + i] = h1[i]; }
        }
        acc0 = __builtin_amdgcn_wmma_f32_16x16x32_f16(false, a0, false, bfrag,
                                                      (short)0, acc0, false, false);
        acc1 = __builtin_amdgcn_wmma_f32_16x16x32_f16(false, a1, false, bfrag,
                                                      (short)0, acc1, false, false);
    }
    float bv = bias[co];
    for (int r8 = 0; r8 < 8; r8++) {
        int m = r8 + hiHalf * 8;
        int p0 = pbase + m;
        if (p0 < HoWo) {
            float v = acc0[r8] + bv;
            v = v > 0.f ? v : 0.2f * v;                 // leaky ReLU
            Y[((long)b * HoWo + p0) * 64 + co] = (half_t)v;
        }
        int p1 = pbase + 16 + m;
        if (p1 < HoWo) {
            float v = acc1[r8] + bv;
            v = v > 0.f ? v : 0.2f * v;
            Y[((long)b * HoWo + p1) * 64 + co] = (half_t)v;
        }
    }
}

// ---------------------------------------------------------------------------
// Global mean pool (19x19) + FC (64 -> fcN), optional tanh01. One block/batch.
// ---------------------------------------------------------------------------
__global__ void poolfc_kernel(const half_t* __restrict__ x4,
                              const float* __restrict__ fc_w,
                              const float* __restrict__ fc_b,
                              float* __restrict__ out, int fcN, int applyTanh) {
    __shared__ float means[64];
    int b = blockIdx.x, c = threadIdx.x;
    float s = 0.f;
    for (int p = 0; p < 361; p++) s += (float)x4[((long)b * 361 + p) * 64 + c];
    means[c] = s * (1.f / 361.f);
    __syncthreads();
    if (c < fcN) {
        float acc = fc_b[c];
        for (int k = 0; k < 64; k++) acc += means[k] * fc_w[k * fcN + c];
        if (applyTanh) acc = 0.5f * tanhf(acc) + 0.5f;
        out[b * fcN + c] = acc;
    }
}

// ---------------------------------------------------------------------------
// Final mask kernel (cubic / ellipse / gradient) -> output f32.
// ---------------------------------------------------------------------------
__device__ __forceinline__ float clampf(float v, float lo, float hi) {
    return fminf(fmaxf(v, lo), hi);
}
__device__ __forceinline__ float tanh01f(float x) { return 0.5f * tanhf(x) + 0.5f; }
__device__ __forceinline__ float sign01f(float x) {
    float s = (x > 0.f) ? 1.f : ((x < 0.f) ? -1.f : 0.f);
    return 0.5f * (s + 1.f);
}

__device__ void ell_group(const float* p, float ax, float ay, float m[3]) {
    float sc0 = p[0] * 2.f + EPS_F, sc1 = p[1] * 2.f + EPS_F, sc2 = p[2] * 2.f + EPS_F;
    float hh = p[3] + EPS_F, kk = p[4] + EPS_F;
    float theta = p[5] * PI_F + EPS_F;
    float a = p[6] + EPS_F, bb = p[7] + EPS_F;
    float dx = ax - hh, dy = ay - kk;
    float r2 = dx * dx + dy * dy;
    float cv = dy / (sqrtf(r2 + EPS_F) + EPS_F);
    cv = clampf(cv, -1.f + 1e-7f, 1.f - 1e-7f);
    float angle = acosf(cv) - theta;
    float sa = sinf(angle), ca = cosf(angle);
    float radius = a * bb / (sqrtf(a * a * sa * sa + bb * bb * ca * ca + EPS_F) + EPS_F) + EPS_F;
    float st = sinf(theta), ct = cosf(theta);
    float t1 = dx * ct + dy * st, t2 = dx * st - dy * ct;
    float cond = t1 * t1 / (a * a) + t2 * t2 / (bb * bb) + EPS_F;
    float mm = sqrtf(r2 + EPS_F);
    float sc[3] = {sc0, sc1, sc2};
    for (int i = 0; i < 3; i++) {
        float v = (cond < 1.f) ? (mm * (1.f - sc[i]) / radius + sc[i]) : 1.f;
        m[i] = clampf(v, 0.f, 2.f);
    }
}

__device__ void grad_group(const float* p, float ax, float ay, float m[3]) {
    float ginv = sign01f(p[7]);
    float slope = atanf(p[3]);
    float c = tanh01f(p[4]) + EPS_F;
    float o1 = clampf(sign01f(p[5]), c, 1.f);
    float o2 = clampf(sign01f(p[6]), 0.f, c);
    float s3[3] = { tanh01f(p[0]) * 2.f, tanh01f(p[1]) * 2.f, tanh01f(p[2]) * 2.f };
    float cs = cosf(slope);
    float d1 = tanh01f(o1) * cs, d2 = tanh01f(o2) * cs;
    float top = tanh01f(ay - (p[3] * ax + c + d1));
    for (int i = 0; i < 3; i++) {
        float s = s3[i];
        float f = (ginv == 0.f) ? ((s - 1.f) / (2.f * d1) + (s - 1.f) / (2.f * d2))
                                : ((s - 1.f) / (2.f * d1) + (1.f - s) / (2.f * d2));
        float gr = (ginv == 0.f) ? s : 1.f;
        float lo = (s >= 1.f) ? 1.f : 0.f;
        float hi = (s >= 1.f) ? 2.f : 1.f;
        m[i] = clampf(clampf(f + gr * top, lo, hi), 0.f, 2.f);
    }
}

__global__ void final_kernel(const float* __restrict__ img,
                             const float* __restrict__ fac_cubic,
                             const float* __restrict__ fac_ell,
                             const float* __restrict__ fac_grad,
                             float* __restrict__ out) {
    long idx = (long)blockIdx.x * blockDim.x + threadIdx.x;
    if (idx >= 4L * 512 * 512) return;
    int j = (int)(idx % 512);
    long r = idx / 512;
    int i = (int)(r % 512);
    int b = (int)(r / 512);
    float ax = (float)i / 512.f, ay = (float)j / 512.f;
    long base = (((long)b * 512 + i) * 512 + j) * 3;
    float R[3] = { img[base], img[base + 1], img[base + 2] };

    float cub[3];
    float axm = ax * ay;
    for (int ci = 0; ci < 3; ci++) {
        const float* kern = fac_cubic + b * 60 + ci * 20;
        float ch[4] = { ax, ay, 1.f, R[ci] };
        float acc = 0.f;
        for (int pp = 0; pp < 4; pp++)
            for (int q = 0; q < 4; q++)
                acc += kern[pp * 4 + q] * ch[pp] * ch[q] * ch[q];
        for (int pp = 0; pp < 4; pp++)
            acc += kern[16 + pp] * axm * R[ci] / (ch[pp] + 1e-8f);
        cub[ci] = acc;
    }

    float e0[3], e1[3], e2[3];
    ell_group(fac_ell + b * 24 + 0,  ax, ay, e0);
    ell_group(fac_ell + b * 24 + 8,  ax, ay, e1);
    ell_group(fac_ell + b * 24 + 16, ax, ay, e2);
    float g0[3], g1[3], g2[3];
    grad_group(fac_grad + b * 24 + 0,  ax, ay, g0);
    grad_group(fac_grad + b * 24 + 8,  ax, ay, g1);
    grad_group(fac_grad + b * 24 + 16, ax, ay, g2);

    for (int ci = 0; ci < 3; ci++) {
        float ell = clampf(e0[ci] * e1[ci] * e2[ci], 0.f, 2.f);
        float gr  = clampf(g0[ci] * g1[ci] * g2[ci], 0.f, 2.f);
        float mask = clampf(gr * ell, 0.f, 2.f);
        out[base + ci] = clampf((R[ci] + cub[ci]) * mask, 0.f, 1.f);
    }
}

// ---------------------------------------------------------------------------
extern "C" void kernel_launch(void* const* d_in, const int* in_sizes, int n_in,
                              void* d_out, int out_size, void* d_ws, size_t ws_size,
                              hipStream_t stream) {
    (void)in_sizes; (void)n_in; (void)out_size; (void)ws_size;
    const float* img    = (const float*)d_in[0];
    const float* bb_w   = (const float*)d_in[1];
    const float* bb_b   = (const float*)d_in[2];
    const float* cub_cw = (const float*)d_in[3];
    const float* cub_cb = (const float*)d_in[4];
    const float* cub_fw = (const float*)d_in[5];
    const float* cub_fb = (const float*)d_in[6];
    const float* ell_cw = (const float*)d_in[7];
    const float* ell_cb = (const float*)d_in[8];
    const float* ell_fw = (const float*)d_in[9];
    const float* ell_fb = (const float*)d_in[10];
    const float* grd_cw = (const float*)d_in[11];
    const float* grd_cb = (const float*)d_in[12];
    const float* grd_fw = (const float*)d_in[13];
    const float* grd_fb = (const float*)d_in[14];

    char* ws = (char*)d_ws;
    size_t off = 0;
    auto alloc = [&](size_t bytes) { size_t o = off; off += (bytes + 255) & ~(size_t)255; return o; };
    half_t* wt_bb = (half_t*)(ws + alloc(64UL * 32 * 2));
    half_t* wt_hd = (half_t*)(ws + alloc(3UL * 4 * 64 * 576 * 2));
    half_t* fi    = (half_t*)(ws + alloc(4UL * 512 * 512 * 64 * 2));
    half_t* x0v   = (half_t*)(ws + alloc(4UL * 300 * 300 * 64 * 2));
    half_t* xa    = (half_t*)(ws + alloc(4UL * 150 * 150 * 64 * 2));
    half_t* xb    = (half_t*)(ws + alloc(4UL * 75 * 75 * 64 * 2));
    float*  fac_c = (float*)(ws + alloc(4UL * 60 * 4));
    float*  fac_e = (float*)(ws + alloc(4UL * 24 * 4));
    float*  fac_g = (float*)(ws + alloc(4UL * 24 * 4));

    auto cdiv = [](long a, long b) { return (a + b - 1) / b; };

    prep_bb_kernel<<<cdiv(64 * 32, 256), 256, 0, stream>>>(bb_w, wt_bb);
    prep_head_kernel<<<cdiv(3L * 4 * 64 * 576, 256), 256, 0, stream>>>(cub_cw, ell_cw, grd_cw, wt_hd);
    backbone_kernel<<<4 * (512 * 512 / 16), 128, 0, stream>>>(img, wt_bb, bb_b, fi);
    resize_kernel<<<cdiv(4L * 300 * 300 * 64, 256), 256, 0, stream>>>(fi, x0v);

    const float* cbs[3]  = { cub_cb, ell_cb, grd_cb };
    const float* fws[3]  = { cub_fw, ell_fw, grd_fw };
    const float* fbs[3]  = { cub_fb, ell_fb, grd_fb };
    float*       facs[3] = { fac_c, fac_e, fac_g };
    int fcNs[3]  = { 60, 24, 24 };
    int tanhs[3] = { 0, 1, 0 };
    int dims[5]  = { 300, 150, 75, 38, 19 };
    int pads[4]  = { 0, 0, 1, 0 };

    for (int h = 0; h < 3; h++) {
        half_t* bufs[5] = { x0v, xa, xb, xa, xb };
        for (int l = 0; l < 4; l++) {
            int Hi = dims[l], Ho = dims[l + 1];
            long grid = 4L * cdiv((long)Ho * Ho, 32);
            conv2s_kernel<<<grid, 128, 0, stream>>>(
                bufs[l], wt_hd + (long)(h * 4 + l) * 64 * 576,
                cbs[h] + l * 64, bufs[l + 1], Hi, Hi, Ho, Ho, pads[l]);
        }
        poolfc_kernel<<<4, 64, 0, stream>>>(xb, fws[h], fbs[h], facs[h], fcNs[h], tanhs[h]);
    }

    final_kernel<<<cdiv(4L * 512 * 512, 256), 256, 0, stream>>>(
        img, fac_c, fac_e, fac_g, (float*)d_out);
}
